// MultiHeadSelfAttention_Flash_11115375362312
// MI455X (gfx1250) — matrix operand order repaired and checked by reading, not compiled, not run on hardware
//
#include <hip/hip_runtime.h>
#include <hip/hip_bf16.h>

typedef __attribute__((ext_vector_type(16))) __bf16 v16bf;
typedef __attribute__((ext_vector_type(8)))  __bf16 v8bf;
typedef __attribute__((ext_vector_type(8)))  float  v8f;
typedef __attribute__((ext_vector_type(4)))  float  v4f;
typedef __attribute__((ext_vector_type(4)))  unsigned int v4u;
typedef __attribute__((ext_vector_type(8)))  int    v8i;
typedef __attribute__((ext_vector_type(4)))  int    v4i;

#define NHEAD  16
#define DQKV   64
#define DMODEL 1024
#define BATCH  8
#define SEQ    1024
#define NTOK   (BATCH * SEQ)
#define QKVDIM (3 * NHEAD * DQKV)   // 3072

#define TILE_PITCH 40               // 32 bf16 + 8 pad -> 80B row pitch in LDS

#if __has_builtin(__builtin_amdgcn_tensor_load_to_lds) && __has_builtin(__builtin_amdgcn_s_wait_tensorcnt)
#define USE_TDM 1
#else
#define USE_TDM 0
#endif

__device__ __forceinline__ v8f wmma_bf16(v16bf a, v16bf b, v8f c) {
    return __builtin_amdgcn_wmma_f32_16x16x32_bf16(false, a, false, b, (short)0, c, false, false);
}

#if USE_TDM
// 2-D TDM: load a tile_d0(=32 contiguous) x tile_d1(=64 rows) bf16 tile into LDS
// with 16B padding after each 64B row (pad_interval=3 -> 16 DWORDs, pad_amount=3 -> 4 DWORDs).
__device__ __forceinline__ void tdm_load_2d(unsigned lds_addr, const __bf16* gaddr,
                                            unsigned tensor_d0, unsigned tensor_d1,
                                            unsigned tile_d0, unsigned tile_d1,
                                            unsigned long long stride0) {
    unsigned long long ga = (unsigned long long)gaddr;
    v4u g0;
    g0[0] = 1u;                                            // count=1 (valid), user mode
    g0[1] = lds_addr;                                      // lds_addr [63:32]
    g0[2] = (unsigned)(ga & 0xffffffffu);                  // global_addr [95:64]
    g0[3] = (unsigned)((ga >> 32) & 0x1ffffffu) | (2u << 30); // global_addr hi + type=2
    v8i g1;
    const unsigned pad_en = 1u, pad_interval = 3u, pad_amount = 3u;
    g1[0] = (int)((1u << 16) | (pad_en << 20) | (pad_interval << 22) | (pad_amount << 25)); // data_size=2B
    g1[1] = (int)(tensor_d0 << 16);                        // tensor_dim0[15:0] (atomic addr = 0)
    g1[2] = (int)((tensor_d0 >> 16) | (tensor_d1 << 16));  // tensor_dim0 hi | tensor_dim1 lo
    g1[3] = (int)((tensor_d1 >> 16) | (tile_d0 << 16));    // tensor_dim1 hi | tile_dim0
    g1[4] = (int)(tile_d1 & 0xffffu);                      // tile_dim1 (tile_dim2 = 0)
    g1[5] = (int)(stride0 & 0xffffffffu);                  // tensor_dim0_stride lo
    g1[6] = (int)((stride0 >> 32) & 0xffffu);              // stride hi (dim1_stride = 0)
    g1[7] = 0;
    v4i g2 = {0, 0, 0, 0};
    v4i g3 = {0, 0, 0, 0};
    v8i g4 = {0, 0, 0, 0, 0, 0, 0, 0};
    __builtin_amdgcn_tensor_load_to_lds(g0, g1, g2, g3, g4, 0);
}
#endif

// Stage a 64-row x 32-col bf16 B tile (rows n0..n0+63, cols k0..k0+31 of wsrc[ntot][DMODEL])
__device__ __forceinline__ void stage_issue(const __bf16* __restrict__ wsrc, __bf16* tile,
                                            int n0, int k0, int ntot, int wave, int tid) {
#if USE_TDM
    if (wave == 0) {
        tdm_load_2d((unsigned)(unsigned long long)(void*)tile,
                    wsrc + (size_t)n0 * DMODEL + k0,
                    DMODEL, (unsigned)ntot, 32u, 64u, DMODEL);
    }
#else
    const int row = tid >> 2, ch = tid & 3;
    v8bf tmp = *(const v8bf*)(wsrc + (size_t)(n0 + row) * DMODEL + k0 + ch * 8);
    *(v8bf*)(tile + row * TILE_PITCH + ch * 8) = tmp;
#endif
}

__device__ __forceinline__ void stage_wait(int wave) {
#if USE_TDM
    if (wave == 0) __builtin_amdgcn_s_wait_tensorcnt(0);
#endif
}

// ---------------------------------------------------------------------------
// Kernel 0: fp32 -> bf16 bulk conversion (x, w_qkv, w_o), done once per call.
// ---------------------------------------------------------------------------
__global__ __launch_bounds__(256) void cvt_bf16(const float* __restrict__ in,
                                                __bf16* __restrict__ out, int n4) {
    const int t = blockIdx.x * 256 + threadIdx.x;
    if (t < n4) {
        v4f v = ((const v4f*)in)[t];
        out[4 * t + 0] = (__bf16)v[0];
        out[4 * t + 1] = (__bf16)v[1];
        out[4 * t + 2] = (__bf16)v[2];
        out[4 * t + 3] = (__bf16)v[3];
    }
}

// ---------------------------------------------------------------------------
// Kernel 1: QKV projection, bf16 operands, TDM/LDS-staged B tiles.
// Block: 256 thr = 8 waves; wave w -> rows [by*128 + w*16, +16), cols [bx*64, +64)
// ---------------------------------------------------------------------------
__global__ __launch_bounds__(256) void qkv_gemm(const __bf16* __restrict__ xb,
                                                const __bf16* __restrict__ wb,
                                                __bf16* __restrict__ qkv) {
    __shared__ __bf16 tileB[2][64 * TILE_PITCH];

    const int tid  = threadIdx.x;
    const int lane = tid & 31;
    const int wave = tid >> 5;
    const int m0 = blockIdx.y * 128 + wave * 16;
    const int n0 = blockIdx.x * 64;
    const int mrow = m0 + (lane & 15);
    const int akb = (lane < 16) ? 0 : 16;
    const int bkk = (lane < 16) ? 0 : 16;
    const int nB  = lane & 15;

    v8f acc[4];
#pragma unroll
    for (int nt = 0; nt < 4; nt++)
#pragma unroll
        for (int r = 0; r < 8; r++) acc[nt][r] = 0.0f;

    stage_issue(wb, &tileB[0][0], n0, 0, QKVDIM, wave, tid);

    for (int i = 0; i < DMODEL / 32; i++) {
        const int k0 = i * 32;
        const int buf = i & 1;
        stage_wait(wave);
        __syncthreads();
        if (i + 1 < DMODEL / 32)
            stage_issue(wb, &tileB[buf ^ 1][0], n0, k0 + 32, QKVDIM, wave, tid);

        const __bf16* ap = xb + (size_t)mrow * DMODEL + k0 + akb;
        v8bf lo = *(const v8bf*)ap;
        v8bf hh = *(const v8bf*)(ap + 8);
        v16bf af;
#pragma unroll
        for (int j = 0; j < 8; j++) { af[j] = lo[j]; af[8 + j] = hh[j]; }

#pragma unroll
        for (int nt = 0; nt < 4; nt++) {
            const __bf16* bp = &tileB[buf][(nt * 16 + nB) * TILE_PITCH + bkk];
            v8bf b0 = *(const v8bf*)bp;
            v8bf b1 = *(const v8bf*)(bp + 8);
            v16bf bf;
#pragma unroll
            for (int j = 0; j < 8; j++) { bf[j] = b0[j]; bf[8 + j] = b1[j]; }
            acc[nt] = wmma_bf16(af, bf, acc[nt]);
        }
    }

    const int hi = lane >> 4;
#pragma unroll
    for (int nt = 0; nt < 4; nt++) {
        const int col = n0 + nt * 16 + nB;
#pragma unroll
        for (int r = 0; r < 8; r++) {
            const int row = m0 + r + 8 * hi;
            qkv[(size_t)row * QKVDIM + col] = (__bf16)acc[nt][r];
        }
    }
}

// ---------------------------------------------------------------------------
// Kernel 2: RoPE on q,k (in place, bf16) + V transpose to vt[b][h][d][s].
// ---------------------------------------------------------------------------
__global__ __launch_bounds__(256) void rope_vt(__bf16* __restrict__ qkv,
                                               __bf16* __restrict__ vt,
                                               const int* __restrict__ seq_lens) {
    const int gid = blockIdx.x * blockDim.x + threadIdx.x;
    if (gid >= NTOK * NHEAD * 32) return;
    const int i    = gid & 31;
    const int head = (gid >> 5) & (NHEAD - 1);
    const int tok  = gid >> 9;

    int rem = tok, pos = 0;
#pragma unroll
    for (int b = 0; b < BATCH; b++) {
        int L = seq_lens[b];
        if (rem < L) { pos = rem; break; }
        rem -= L;
    }

    const int j0 = (2 * i) & 31;
    const int j1 = (2 * i + 1) & 31;
    const float lnb = 9.210340371976184f;   // ln(10000)
    const float f0 = __expf(-(float)j0 * (1.0f / 32.0f) * lnb);
    const float f1 = __expf(-(float)j1 * (1.0f / 32.0f) * lnb);
    float s0, c0, s1, c1;
    __sincosf((float)pos * f0, &s0, &c0);
    __sincosf((float)pos * f1, &s1, &c1);

#pragma unroll
    for (int blk = 0; blk < 2; blk++) {   // 0 = q, 1 = k
        __bf16* p = qkv + (size_t)tok * QKVDIM + blk * (NHEAD * DQKV) + head * DQKV + 2 * i;
        const float e = (float)p[0];
        const float o = (float)p[1];
        p[0] = (__bf16)(e * c0 - o * s0);
        p[1] = (__bf16)(o * c1 + e * s1);
    }

    const int b  = tok >> 10;
    const int sl = tok & (SEQ - 1);
    const __bf16* vp = qkv + (size_t)tok * QKVDIM + 2 * (NHEAD * DQKV) + head * DQKV + 2 * i;
    __bf16* vo = vt + (((size_t)(b * NHEAD + head) * DQKV + 2 * i) * SEQ) + sl;
    vo[0]   = vp[0];
    vo[SEQ] = vp[1];
}

// ---------------------------------------------------------------------------
// Kernel 3: flash attention. Block = 4 waves, each wave owns 16 query rows.
// ---------------------------------------------------------------------------
__global__ __launch_bounds__(128) void flash_attn(const __bf16* __restrict__ qkv,
                                                  const __bf16* __restrict__ vt,
                                                  __bf16* __restrict__ attn_out) {
    __shared__ __bf16 lds_p[4][16][32];

    const int lane  = threadIdx.x & 31;
    const int wave  = threadIdx.x >> 5;
    const int batch = blockIdx.z;
    const int head  = blockIdx.y;
    const int q0    = blockIdx.x * 64 + wave * 16;
    const int tokBase = batch * SEQ;

    const int mA  = lane & 15;
    const int kbA = (lane < 16) ? 0 : 16;
    const int kkB = (lane < 16) ? 0 : 16;
    const int nB  = lane & 15;
    const int hi  = lane >> 4;

    v16bf qa[2];
    const __bf16* qbase = qkv + (size_t)(tokBase + q0 + mA) * QKVDIM + head * DQKV;
#pragma unroll
    for (int ks = 0; ks < 2; ks++) {
        v8bf lo = *(const v8bf*)(qbase + ks * 32 + kbA);
        v8bf hh = *(const v8bf*)(qbase + ks * 32 + kbA + 8);
#pragma unroll
        for (int j = 0; j < 8; j++) { qa[ks][j] = lo[j]; qa[ks][8 + j] = hh[j]; }
    }

    float mrow[8], lrow[8];
    v8f o[4];
#pragma unroll
    for (int r = 0; r < 8; r++) { mrow[r] = -1e30f; lrow[r] = 0.0f; }
#pragma unroll
    for (int nt = 0; nt < 4; nt++)
#pragma unroll
        for (int r = 0; r < 8; r++) o[nt][r] = 0.0f;

    const __bf16* kbase = qkv + (size_t)tokBase * QKVDIM + (NHEAD * DQKV) + head * DQKV;
    const __bf16* vbase = vt + (size_t)(batch * NHEAD + head) * DQKV * SEQ;
    const float scale = 0.125f;

    for (int kc = 0; kc < SEQ; kc += 32) {
        v8f s0, s1;
#pragma unroll
        for (int r = 0; r < 8; r++) { s0[r] = 0.0f; s1[r] = 0.0f; }
#pragma unroll
        for (int ks = 0; ks < 2; ks++) {
            const __bf16* kp0 = kbase + (size_t)(kc + nB) * QKVDIM + ks * 32 + kkB;
            const __bf16* kp1 = kbase + (size_t)(kc + 16 + nB) * QKVDIM + ks * 32 + kkB;
            v16bf b0 = *(const v16bf*)kp0;
            v16bf b1 = *(const v16bf*)kp1;
            s0 = wmma_bf16(qa[ks], b0, s0);
            s1 = wmma_bf16(qa[ks], b1, s1);
        }

        float alpha[8];
#pragma unroll
        for (int r = 0; r < 8; r++) {
            s0[r] *= scale; s1[r] *= scale;
            float t = fmaxf(s0[r], s1[r]);
#pragma unroll
            for (int xm = 1; xm < 16; xm <<= 1) t = fmaxf(t, __shfl_xor(t, xm, 32));
            const float mn = fmaxf(mrow[r], t);
            alpha[r] = __expf(mrow[r] - mn);
            s0[r] = __expf(s0[r] - mn);
            s1[r] = __expf(s1[r] - mn);
            float u = s0[r] + s1[r];
#pragma unroll
            for (int xm = 1; xm < 16; xm <<= 1) u += __shfl_xor(u, xm, 32);
            lrow[r] = lrow[r] * alpha[r] + u;
            mrow[r] = mn;
        }
#pragma unroll
        for (int nt = 0; nt < 4; nt++)
#pragma unroll
            for (int r = 0; r < 8; r++) o[nt][r] *= alpha[r];

#pragma unroll
        for (int r = 0; r < 8; r++) {
            lds_p[wave][r + 8 * hi][nB]      = (__bf16)s0[r];
            lds_p[wave][r + 8 * hi][nB + 16] = (__bf16)s1[r];
        }
        __syncthreads();
        v16bf pa;
        {
            v8bf lo = *(const v8bf*)&lds_p[wave][mA][kbA];
            v8bf hh = *(const v8bf*)&lds_p[wave][mA][kbA + 8];
#pragma unroll
            for (int j = 0; j < 8; j++) { pa[j] = lo[j]; pa[8 + j] = hh[j]; }
        }
        __syncthreads();

#pragma unroll
        for (int nt = 0; nt < 4; nt++) {
            const __bf16* vp = vbase + (size_t)(nt * 16 + nB) * SEQ + kc + kkB;
            v16bf vfrag = *(const v16bf*)vp;
            o[nt] = wmma_bf16(pa, vfrag, o[nt]);
        }
    }

#pragma unroll
    for (int nt = 0; nt < 4; nt++) {
#pragma unroll
        for (int r = 0; r < 8; r++) {
            const float val = o[nt][r] / lrow[r];
            const int row = tokBase + q0 + r + 8 * hi;
            attn_out[(size_t)row * (NHEAD * DQKV) + head * DQKV + nt * 16 + nB] = (__bf16)val;
        }
    }
}

// ---------------------------------------------------------------------------
// Kernel 4: output projection + residual, bf16 operands, TDM/LDS-staged B.
// ---------------------------------------------------------------------------
__global__ __launch_bounds__(256) void out_proj(const __bf16* __restrict__ a,
                                                const __bf16* __restrict__ wob,
                                                const float* __restrict__ x,
                                                float* __restrict__ out) {
    __shared__ __bf16 tileB[2][64 * TILE_PITCH];

    const int tid  = threadIdx.x;
    const int lane = tid & 31;
    const int wave = tid >> 5;
    const int m0 = blockIdx.y * 128 + wave * 16;
    const int n0 = blockIdx.x * 64;
    const int mrow = m0 + (lane & 15);
    const int akb = (lane < 16) ? 0 : 16;
    const int bkk = (lane < 16) ? 0 : 16;
    const int nB  = lane & 15;

    v8f acc[4];
#pragma unroll
    for (int nt = 0; nt < 4; nt++)
#pragma unroll
        for (int r = 0; r < 8; r++) acc[nt][r] = 0.0f;

    stage_issue(wob, &tileB[0][0], n0, 0, DMODEL, wave, tid);

    for (int i = 0; i < DMODEL / 32; i++) {
        const int k0 = i * 32;
        const int buf = i & 1;
        stage_wait(wave);
        __syncthreads();
        if (i + 1 < DMODEL / 32)
            stage_issue(wob, &tileB[buf ^ 1][0], n0, k0 + 32, DMODEL, wave, tid);

        const __bf16* ap = a + (size_t)mrow * DMODEL + k0 + akb;
        v8bf lo = *(const v8bf*)ap;
        v8bf hh = *(const v8bf*)(ap + 8);
        v16bf af;
#pragma unroll
        for (int j = 0; j < 8; j++) { af[j] = lo[j]; af[8 + j] = hh[j]; }

#pragma unroll
        for (int nt = 0; nt < 4; nt++) {
            const __bf16* bp = &tileB[buf][(nt * 16 + nB) * TILE_PITCH + bkk];
            v8bf b0 = *(const v8bf*)bp;
            v8bf b1 = *(const v8bf*)(bp + 8);
            v16bf bf;
#pragma unroll
            for (int j = 0; j < 8; j++) { bf[j] = b0[j]; bf[8 + j] = b1[j]; }
            acc[nt] = wmma_bf16(af, bf, acc[nt]);
        }
    }

    const int hi = lane >> 4;
#pragma unroll
    for (int nt = 0; nt < 4; nt++) {
        const int col = n0 + nt * 16 + nB;
#pragma unroll
        for (int r = 0; r < 8; r++) {
            const int row = m0 + r + 8 * hi;
            out[(size_t)row * DMODEL + col] = acc[nt][r] + x[(size_t)row * DMODEL + col];
        }
    }
}

// ---------------------------------------------------------------------------
// Kernel 5: LayerNorm in place on d_out. One block (256 thr) per row of 1024.
// ---------------------------------------------------------------------------
__global__ __launch_bounds__(256) void layernorm(float* __restrict__ out,
                                                 const float* __restrict__ gamma,
                                                 const float* __restrict__ beta) {
    __shared__ float red_s[8];
    __shared__ float red_q[8];
    const int row = blockIdx.x;
    float* p = out + (size_t)row * DMODEL;

    float v[4];
    float s = 0.0f, q = 0.0f;
#pragma unroll
    for (int j = 0; j < 4; j++) {
        v[j] = p[threadIdx.x + j * 256];
        s += v[j];
        q += v[j] * v[j];
    }
#pragma unroll
    for (int xm = 1; xm < 32; xm <<= 1) {
        s += __shfl_xor(s, xm, 32);
        q += __shfl_xor(q, xm, 32);
    }
    const int lane = threadIdx.x & 31, wave = threadIdx.x >> 5;
    if (lane == 0) { red_s[wave] = s; red_q[wave] = q; }
    __syncthreads();
    s = 0.0f; q = 0.0f;
#pragma unroll
    for (int wv = 0; wv < 8; wv++) { s += red_s[wv]; q += red_q[wv]; }

    const float mean = s * (1.0f / DMODEL);
    const float var  = q * (1.0f / DMODEL) - mean * mean;
    const float rstd = rsqrtf(var + 1e-6f);
#pragma unroll
    for (int j = 0; j < 4; j++) {
        const int c = threadIdx.x + j * 256;
        p[c] = (v[j] - mean) * rstd * gamma[c] + beta[c];
    }
}

// ---------------------------------------------------------------------------
extern "C" void kernel_launch(void* const* d_in, const int* in_sizes, int n_in,
                              void* d_out, int out_size, void* d_ws, size_t ws_size,
                              hipStream_t stream) {
    const float* x        = (const float*)d_in[0];
    const int*   seq_lens = (const int*)d_in[1];
    const float* w_qkv    = (const float*)d_in[2];
    const float* w_o      = (const float*)d_in[3];
    const float* gamma    = (const float*)d_in[4];
    const float* beta     = (const float*)d_in[5];
    float* out = (float*)d_out;

    char* ws = (char*)d_ws;
    size_t off = 0;
    __bf16* qkv   = (__bf16*)(ws + off); off += (size_t)NTOK * QKVDIM * 2;           // 48 MB
    __bf16* vt    = (__bf16*)(ws + off); off += (size_t)BATCH * NHEAD * DQKV * SEQ * 2; // 16 MB
    __bf16* attn  = (__bf16*)(ws + off); off += (size_t)NTOK * DMODEL * 2;           // 16 MB
    __bf16* x_bf  = (__bf16*)(ws + off); off += (size_t)NTOK * DMODEL * 2;           // 16 MB
    __bf16* wq_bf = (__bf16*)(ws + off); off += (size_t)QKVDIM * DMODEL * 2;         // 6 MB
    __bf16* wo_bf = (__bf16*)(ws + off); off += (size_t)DMODEL * DMODEL * 2;         // 2 MB

    cvt_bf16<<<(NTOK * DMODEL / 4 + 255) / 256, 256, 0, stream>>>(x, x_bf, NTOK * DMODEL / 4);
    cvt_bf16<<<(QKVDIM * DMODEL / 4 + 255) / 256, 256, 0, stream>>>(w_qkv, wq_bf, QKVDIM * DMODEL / 4);
    cvt_bf16<<<(DMODEL * DMODEL / 4 + 255) / 256, 256, 0, stream>>>(w_o, wo_bf, DMODEL * DMODEL / 4);

    qkv_gemm<<<dim3(QKVDIM / 64, NTOK / 128), 256, 0, stream>>>(x_bf, wq_bf, qkv);
    rope_vt<<<(NTOK * NHEAD * 32) / 256, 256, 0, stream>>>(qkv, vt, seq_lens);
    flash_attn<<<dim3(SEQ / 64, NHEAD, BATCH), 128, 0, stream>>>(qkv, vt, attn);
    out_proj<<<dim3(DMODEL / 64, NTOK / 128), 256, 0, stream>>>(attn, wo_bf, x, out);
    layernorm<<<NTOK, 256, 0, stream>>>(out, gamma, beta);
}
